// MultiHeadAtt_68710886801681
// MI455X (gfx1250) — matrix operand order repaired and checked by reading, not compiled, not run on hardware
//
#include <hip/hip_runtime.h>
#include <hip/hip_bf16.h>

// MHA forward, CDNA5 (gfx1250), wave32.
// v_wmma_f32_16x16x32_bf16 for both GEMM stages + TDM (tensor_load_to_lds)
// double-buffered K-tile staging in the flash-attention kernel.
// B=2, M=4096, D=512, H=8, DK=64.

#define B_  2
#define M_  4096
#define D_  512
#define H_  8
#define DK_ 64

typedef __attribute__((ext_vector_type(16))) __bf16 v16bf;
typedef __attribute__((ext_vector_type(8)))  float  v8f;
typedef __attribute__((ext_vector_type(4)))  unsigned int u32x4;
typedef __attribute__((ext_vector_type(8)))  int          i32x8;
typedef __attribute__((ext_vector_type(4)))  int          i32x4;

union FragB { v16bf v; unsigned int u[8]; };

__device__ __forceinline__ unsigned short f2bf(float f) {
  unsigned int u = __float_as_uint(f);
  u += 0x7FFFu + ((u >> 16) & 1u);      // round-to-nearest-even
  return (unsigned short)(u >> 16);
}
__device__ __forceinline__ unsigned int pack2bf(float a, float b) {
  return (unsigned int)f2bf(a) | ((unsigned int)f2bf(b) << 16);
}
// A-matrix (16-bit, 16x32) pair-p K offset within a lane-half (contiguous: k = 16*half + 2p)
__device__ __forceinline__ int a_koff(int p) {
  return 2 * p;
}

// ---------------------------------------------------------------------------
// TDM: DMA a 2D tile of 16-bit elements (tile_d0 x tile_d1, row stride
// stride0 elements) from global memory into LDS.  D# layout per CDNA5 ISA
// 8.3/8.4.  Issue from ONE wave (uniform branch); tracked by TENSORcnt.
// This toolchain exposes the 6-arg builtin:
//   (u32x4 g0, i32x8 g1, i32x4 g2, i32x4 g3, i32x8 extra, i32 cpol)
// Groups 2/3 + extra are zero (2D tile: tile_dim2 = 0 -> unused dims).
// ---------------------------------------------------------------------------
__device__ __forceinline__ void tdm_load_tile_bf16(
    unsigned lds_byte_addr, const unsigned short* gptr,
    unsigned tile_d0, unsigned tile_d1,
    unsigned tensor_d0, unsigned tensor_d1, unsigned stride0)
{
  unsigned long long ga = (unsigned long long)(size_t)gptr;
  u32x4 g0;
  g0[0] = 1u;                                              // count=1 (valid), user mode
  g0[1] = lds_byte_addr;                                   // lds_addr [63:32]
  g0[2] = (unsigned)(ga & 0xFFFFFFFFu);                    // global_addr [95:64]
  g0[3] = (unsigned)((ga >> 32) & 0x01FFFFFFu)             // global_addr [120:96]
        | (2u << 30);                                      // type=2 ("image")
  i32x8 g1;
  g1[0] = (int)(1u << 16);                                 // wg_mask=0, data_size=1 (2B)
  g1[1] = (int)((tensor_d0 & 0xFFFFu) << 16);              // tensor_dim0[15:0]
  g1[2] = (int)((tensor_d0 >> 16) | ((tensor_d1 & 0xFFFFu) << 16));
  g1[3] = (int)((tensor_d1 >> 16) | (tile_d0 << 16));      // tile_dim0
  g1[4] = (int)tile_d1;                                    // tile_dim1 | tile_dim2=0
  g1[5] = (int)stride0;                                    // tensor_dim0_stride[31:0]
  g1[6] = 0;                                               // stride0[47:32] | stride1[15:0]
  g1[7] = 0;
  i32x4 gz4 = {0, 0, 0, 0};                                // groups 2/3 unused (2D)
  i32x8 gz8 = {0, 0, 0, 0, 0, 0, 0, 0};
  __builtin_amdgcn_tensor_load_to_lds(g0, g1, gz4, gz4, gz8, 0);
}

__device__ __forceinline__ unsigned lds_offset_of(const void* p) {
  // Flat addresses in the LDS aperture carry the LDS byte offset in [31:0].
  return (unsigned)(size_t)p;
}

// ---------------------------------------------------------------------------
// Kernel 1: QKV projections.  C[g][n] = sum_k X[g][k] * W[n][k] + bias[n]
// X fp32 [B*M, D], W fp32 [D, D] (row n = output feature), out bf16 [B*M, D].
// blockIdx.z selects q/k/v.  Block = 128 thr = 4 waves; 64x64 output tile.
// ---------------------------------------------------------------------------
__global__ __launch_bounds__(128) void proj_qkv_kernel(
    const float* __restrict__ Xq, const float* __restrict__ Xk, const float* __restrict__ Xv,
    const float* __restrict__ Wq, const float* __restrict__ bq,
    const float* __restrict__ Wk, const float* __restrict__ bk,
    const float* __restrict__ Wv, const float* __restrict__ bv,
    unsigned short* __restrict__ qws, unsigned short* __restrict__ kws,
    unsigned short* __restrict__ vws)
{
  __shared__ __align__(16) unsigned short ldsX[64 * 64];
  __shared__ __align__(16) unsigned short ldsW[64 * 64];

  const float* X; const float* W; const float* bias; unsigned short* O;
  if (blockIdx.z == 0)      { X = Xq; W = Wq; bias = bq; O = qws; }
  else if (blockIdx.z == 1) { X = Xk; W = Wk; bias = bk; O = kws; }
  else                      { X = Xv; W = Wv; bias = bv; O = vws; }

  const int tid  = threadIdx.x;
  const int wave = tid >> 5;
  const int lane = tid & 31;
  const int half = lane >> 4;
  const int l16  = lane & 15;

  const int row0 = blockIdx.x * 64;   // over flattened B*M
  const int n0   = blockIdx.y * 64;   // output feature tile

  v8f zero = {};
  v8f acc[4];
  #pragma unroll
  for (int t = 0; t < 4; ++t) acc[t] = zero;

  const int srow = tid >> 1;          // staging row 0..63
  const int scol = (tid & 1) * 32;    // staging col base

  for (int kk = 0; kk < D_; kk += 64) {
    __syncthreads();
    { // stage X tile -> bf16 LDS [row][k]
      const float4* src = (const float4*)(X + (size_t)(row0 + srow) * D_ + kk + scol);
      #pragma unroll
      for (int j = 0; j < 8; ++j) {
        float4 f = src[j];
        *(unsigned int*)(ldsX + srow * 64 + scol + j * 4)     = pack2bf(f.x, f.y);
        *(unsigned int*)(ldsX + srow * 64 + scol + j * 4 + 2) = pack2bf(f.z, f.w);
      }
    }
    { // stage W tile -> bf16 LDS [n][k]
      const float4* src = (const float4*)(W + (size_t)(n0 + srow) * D_ + kk + scol);
      #pragma unroll
      for (int j = 0; j < 8; ++j) {
        float4 f = src[j];
        *(unsigned int*)(ldsW + srow * 64 + scol + j * 4)     = pack2bf(f.x, f.y);
        *(unsigned int*)(ldsW + srow * 64 + scol + j * 4 + 2) = pack2bf(f.z, f.w);
      }
    }
    __syncthreads();

    FragB a[2];
    #pragma unroll
    for (int ks = 0; ks < 2; ++ks)
      #pragma unroll
      for (int p = 0; p < 8; ++p)
        a[ks].u[p] = *(const unsigned int*)(
            ldsX + (wave * 16 + l16) * 64 + ks * 32 + half * 16 + a_koff(p));

    #pragma unroll
    for (int t = 0; t < 4; ++t) {
      #pragma unroll
      for (int ks = 0; ks < 2; ++ks) {
        FragB bmat;
        #pragma unroll
        for (int p = 0; p < 8; ++p)
          bmat.u[p] = *(const unsigned int*)(
              ldsW + (t * 16 + l16) * 64 + ks * 32 + half * 16 + 2 * p);
        acc[t] = __builtin_amdgcn_wmma_f32_16x16x32_bf16(
            false, a[ks].v, false, bmat.v, (short)0, acc[t], false, false);
      }
    }
  }

  // bias add + bf16 store (C layout: row = half*8+r, col = t*16 + l16)
  #pragma unroll
  for (int t = 0; t < 4; ++t) {
    const int n  = n0 + t * 16 + l16;
    const float bb = bias[n];
    #pragma unroll
    for (int r = 0; r < 8; ++r) {
      const int g = row0 + wave * 16 + half * 8 + r;
      O[(size_t)g * D_ + n] = f2bf(acc[t][r] + bb);
    }
  }
}

// ---------------------------------------------------------------------------
// Kernel 2: flash attention per (b, h, 64-row q tile).
// Block = 128 thr = 4 waves; wave w owns q rows [w*16, w*16+16).
// K tiles arrive via TDM (double buffered); V tiles staged transposed by the
// waves; online softmax in fp32; S and O matmuls on bf16 WMMA.
// ---------------------------------------------------------------------------
__global__ __launch_bounds__(128) void flash_attn_kernel(
    const unsigned short* __restrict__ Q,
    const unsigned short* __restrict__ K,
    const unsigned short* __restrict__ V,
    const unsigned char* __restrict__ key_mask,
    const unsigned char* __restrict__ query_mask,
    float* __restrict__ out)
{
  __shared__ __align__(16) unsigned short ldsK0[64 * 64];   // [token][feat]
  __shared__ __align__(16) unsigned short ldsK1[64 * 64];   // [token][feat]
  __shared__ __align__(16) unsigned short ldsVt[64 * 64];   // [feat][token]
  __shared__ __align__(16) unsigned short ldsP [4][16 * 64];// per-wave P tile

  const int tid  = threadIdx.x;
  const int wave = tid >> 5;
  const int lane = tid & 31;
  const int half = lane >> 4;
  const int l16  = lane & 15;

  const int qt = blockIdx.x;
  const int h  = blockIdx.y;
  const int b  = blockIdx.z;
  const int q0 = qt * 64;
  const size_t bhbase = (size_t)b * M_ * D_ + (size_t)h * DK_;
  const int NT = M_ / 64;

  // Q A-fragments (row = q0 + wave*16 + l16; K dim = 64 features -> 2 frags)
  FragB aq[2];
  {
    const unsigned short* qrow = Q + bhbase + (size_t)(q0 + wave * 16 + l16) * D_;
    #pragma unroll
    for (int ks = 0; ks < 2; ++ks)
      #pragma unroll
      for (int p = 0; p < 8; ++p)
        aq[ks].u[p] = *(const unsigned int*)(qrow + ks * 32 + half * 16 + a_koff(p));
  }

  bool qv[8];
  #pragma unroll
  for (int r = 0; r < 8; ++r)
    qv[r] = query_mask[b * M_ + q0 + wave * 16 + half * 8 + r] != 0;

  v8f zero = {};
  float mrun[8], lrun[8];
  v8f o[4];
  #pragma unroll
  for (int r = 0; r < 8; ++r) { mrun[r] = -INFINITY; lrun[r] = 0.0f; }
  #pragma unroll
  for (int t = 0; t < 4; ++t) o[t] = zero;

  const float inv_d  = 1.0f / (float)D_;
  const float masked = -1e15f * (1.0f / (float)D_);

  const int srow = tid >> 1;
  const int scol = (tid & 1) * 32;

  const unsigned ldsK0_off = lds_offset_of(ldsK0);
  const unsigned ldsK1_off = lds_offset_of(ldsK1);

  // prologue: TDM fetch of K tile 0 into buffer 0 (wave 0 only, uniform)
  if (wave == 0) {
    tdm_load_tile_bf16(ldsK0_off, K + bhbase, 64, 64, 64, M_, D_);
  }

  for (int kt = 0; kt < NT; ++kt) {
    const int kv0 = kt * 64;
    __syncthreads();   // previous tile's consumers done; buffers reusable

    if (wave == 0) {
      if (kt + 1 < NT) {
        // issue next K tile into the other buffer, then wait for current
        tdm_load_tile_bf16(((kt + 1) & 1) ? ldsK1_off : ldsK0_off,
                           K + bhbase + (size_t)(kv0 + 64) * D_,
                           64, 64, 64, M_, D_);
        __builtin_amdgcn_s_wait_tensorcnt(1);
      } else {
        __builtin_amdgcn_s_wait_tensorcnt(0);
      }
    }

    { // stage V tile transposed -> ldsVt[feat][token]  (b128 global reads)
      const unsigned short* src = V + bhbase + (size_t)(kv0 + srow) * D_ + scol;
      union { uint4 q[4]; unsigned short hh[32]; } vb;
      #pragma unroll
      for (int j = 0; j < 4; ++j) vb.q[j] = *(const uint4*)(src + j * 8);
      #pragma unroll
      for (int j = 0; j < 32; ++j) ldsVt[(scol + j) * 64 + srow] = vb.hh[j];
    }
    if (kt + 1 < NT) { // prefetch next V tile (global_prefetch_b8)
      __builtin_prefetch(V + bhbase + (size_t)(kv0 + 64 + srow) * D_ + scol, 0, 0);
    }
    __syncthreads();

    const unsigned short* kbuf = (kt & 1) ? ldsK1 : ldsK0;

    // S[16 x 64] = Q_wave * K_tile^T
    v8f s[4];
    #pragma unroll
    for (int t = 0; t < 4; ++t) {
      s[t] = zero;
      #pragma unroll
      for (int ks = 0; ks < 2; ++ks) {
        FragB bk_;
        #pragma unroll
        for (int p = 0; p < 8; ++p)
          bk_.u[p] = *(const unsigned int*)(
              kbuf + (t * 16 + l16) * 64 + ks * 32 + half * 16 + 2 * p);
        s[t] = __builtin_amdgcn_wmma_f32_16x16x32_bf16(
            false, aq[ks].v, false, bk_.v, (short)0, s[t], false, false);
      }
    }

    bool km[4];
    #pragma unroll
    for (int t = 0; t < 4; ++t)
      km[t] = key_mask[b * M_ + kv0 + t * 16 + l16] != 0;

    // online softmax (row m lives in lanes {half*16 .. half*16+15})
    float alpha[8];
    #pragma unroll
    for (int r = 0; r < 8; ++r) {
      float vals[4];
      float rmax = -INFINITY;
      #pragma unroll
      for (int t = 0; t < 4; ++t) {
        float x = (km[t] && qv[r]) ? s[t][r] * inv_d : masked;
        vals[t] = x;
        rmax = fmaxf(rmax, x);
      }
      #pragma unroll
      for (int off = 1; off < 16; off <<= 1)
        rmax = fmaxf(rmax, __shfl_xor(rmax, off, 32));
      const float mnew = fmaxf(mrun[r], rmax);
      alpha[r] = __expf(mrun[r] - mnew);
      float psum = 0.0f;
      #pragma unroll
      for (int t = 0; t < 4; ++t) {
        float p = __expf(vals[t] - mnew);
        psum += p;
        ldsP[wave][(half * 8 + r) * 64 + t * 16 + l16] = f2bf(p);
      }
      #pragma unroll
      for (int off = 1; off < 16; off <<= 1)
        psum += __shfl_xor(psum, off, 32);
      lrun[r] = lrun[r] * alpha[r] + psum;
      mrun[r] = mnew;
    }

    #pragma unroll
    for (int t = 0; t < 4; ++t)
      #pragma unroll
      for (int r = 0; r < 8; ++r)
        o[t][r] *= alpha[r];

    // P (C layout) -> A-fragments via wave-private LDS (in-order per wave)
    FragB pf[2];
    #pragma unroll
    for (int f = 0; f < 2; ++f)
      #pragma unroll
      for (int p = 0; p < 8; ++p)
        pf[f].u[p] = *(const unsigned int*)(
            &ldsP[wave][l16 * 64 + f * 32 + half * 16 + a_koff(p)]);

    // O[16 x 64] += P * V_tile  (B from ldsVt: pairs over kv contiguous)
    #pragma unroll
    for (int t = 0; t < 4; ++t) {
      #pragma unroll
      for (int f = 0; f < 2; ++f) {
        FragB bv_;
        #pragma unroll
        for (int p = 0; p < 8; ++p)
          bv_.u[p] = *(const unsigned int*)(
              ldsVt + (t * 16 + l16) * 64 + f * 32 + half * 16 + 2 * p);
        o[t] = __builtin_amdgcn_wmma_f32_16x16x32_bf16(
            false, pf[f].v, false, bv_.v, (short)0, o[t], false, false);
      }
    }
  }

  // normalize and write fp32 output [B, M, D]
  #pragma unroll
  for (int t = 0; t < 4; ++t) {
    #pragma unroll
    for (int r = 0; r < 8; ++r) {
      const int q = q0 + wave * 16 + half * 8 + r;
      const int f = h * DK_ + t * 16 + l16;
      out[((size_t)b * M_ + q) * D_ + f] = o[t][r] / lrun[r];
    }
  }
}

// ---------------------------------------------------------------------------
extern "C" void kernel_launch(void* const* d_in, const int* in_sizes, int n_in,
                              void* d_out, int out_size, void* d_ws, size_t ws_size,
                              hipStream_t stream) {
  const float* key_in   = (const float*)d_in[0];
  const float* query_in = (const float*)d_in[1];
  const float* value_in = (const float*)d_in[2];
  const float* Wq = (const float*)d_in[3];
  const float* bq = (const float*)d_in[4];
  const float* bk = (const float*)d_in[6];
  const float* Wk = (const float*)d_in[5];
  const float* Wv = (const float*)d_in[7];
  const float* bv = (const float*)d_in[8];
  const unsigned char* key_mask   = (const unsigned char*)d_in[9];
  const unsigned char* query_mask = (const unsigned char*)d_in[10];
  float* out = (float*)d_out;

  // bf16 workspace: Q | K | V, each B*M*D elements (24 MB total)
  const size_t elems = (size_t)B_ * M_ * D_;
  unsigned short* qws = (unsigned short*)d_ws;
  unsigned short* kws = qws + elems;
  unsigned short* vws = kws + elems;

  dim3 g1((B_ * M_) / 64, D_ / 64, 3);
  proj_qkv_kernel<<<g1, 128, 0, stream>>>(query_in, key_in, value_in,
                                          Wq, bq, Wk, bk, Wv, bv,
                                          qws, kws, vws);

  dim3 g2(M_ / 64, H_, B_);
  flash_attn_kernel<<<g2, 128, 0, stream>>>(qws, kws, vws,
                                            key_mask, query_mask, out);
}
